// Model_463856468346
// MI455X (gfx1250) — compile-verified, matrix operand order checked by reading, not run on hardware
//
#include <hip/hip_runtime.h>
#include <math.h>

// ---------------- problem constants (match reference) ----------------
#define DD      128
#define BGR     8
#define NNODES  32768
#define NEDGES  524288
#define NPAIRS  (NNODES + NEDGES)   // 557056

#define ASTRIDE 136   // padded bf16 row stride for LDS A / KV tiles (272B, 16B-aligned, conflict-free)
#define TSTRIDE 40    // padded bf16 row stride for transposed KV / P tiles (80B)

typedef __bf16 bf16_t;
typedef __bf16 v16bf __attribute__((ext_vector_type(16)));
typedef __bf16 v8bf  __attribute__((ext_vector_type(8)));
typedef __bf16 v4bf  __attribute__((ext_vector_type(4)));
typedef float  v8f   __attribute__((ext_vector_type(8)));

// Per-lane 16-bit WMMA fragment = two contiguous 8-element K-runs:
//   k in [8*half, 8*half+7]  and  [16+8*half, 23+8*half]
__device__ __forceinline__ v16bf cat8(v8bf lo, v8bf hi) {
  return __builtin_shufflevector(lo, hi, 0,1,2,3,4,5,6,7,8,9,10,11,12,13,14,15);
}
__device__ __forceinline__ float lrelu_f(float v) { return v >= 0.f ? v : 0.2f * v; }
__device__ __forceinline__ float sigm_f(float v)  { return 1.f / (1.f + __expf(-v)); }

// ---------------- weight pack: f32 [128x128] -> bf16 B-fragment order ----------------
// P[(((kb*8 + c)*32 + lane)*16 + i]  holds W[k][c*16+n] with
//   half = lane>>4, n = lane&15, g = i>>1, lob = i&1,
//   k = kb*32 + ((g<4)? 2g : 16+2(g-4)) + 8*half + lob
__global__ void k_packW(const float* __restrict__ W, bf16_t* __restrict__ P) {
  int idx = blockIdx.x * 256 + threadIdx.x;
  if (idx >= DD * DD) return;
  const int i  = idx & 15;
  const int l  = (idx >> 4) & 31;
  const int cc = (idx >> 9) & 7;
  const int kb = idx >> 12;
  const int half = l >> 4, n = l & 15;
  const int g = i >> 1, lob = i & 1;
  const int k = kb * 32 + (((g < 4) ? 2 * g : 16 + 2 * (g - 4)) + 8 * half) + lob;
  P[idx] = (bf16_t)W[k * DD + cc * 16 + n];
}

// ---------------- fills ----------------
__global__ void k_fill_f32(float* p, float v, int n) {
  int i = blockIdx.x * 256 + threadIdx.x;
  if (i < n) p[i] = v;
}
__global__ void k_fill_i32(int* p, int v, int n) {
  int i = blockIdx.x * 256 + threadIdx.x;
  if (i < n) p[i] = v;
}

// ---------------- GEMM: C[M x 128] = act(A[M x 128] @ W[128 x 128] + bias) ----
// Wp is pre-packed bf16 (k_packW). Block = 128 threads (4 waves), 64 rows/block.
__global__ __launch_bounds__(128) void k_gemm128(
    const float* __restrict__ A, const bf16_t* __restrict__ Wp,
    const float* __restrict__ bias, float* __restrict__ C, int M, int act)
{
  __shared__ __align__(16) bf16_t sA[64 * ASTRIDE];   // ~17 KB bf16 A tile
  const int tid = threadIdx.x;
  const long rowbase = (long)blockIdx.x * 64;
  const float* Ab = A + rowbase * DD;
  for (int i = tid; i < 64 * DD / 4; i += 128) {
    const float4 v = ((const float4*)Ab)[i];
    const int r = (i * 4) >> 7, col = (i * 4) & 127;
    v4bf pk;
    pk[0] = (bf16_t)v.x; pk[1] = (bf16_t)v.y; pk[2] = (bf16_t)v.z; pk[3] = (bf16_t)v.w;
    *(v4bf*)&sA[r * ASTRIDE + col] = pk;
  }
  __syncthreads();

  const int wave = tid >> 5, lane = tid & 31, half = lane >> 4, m = lane & 15;
  const v16bf* wfrag = (const v16bf*)Wp;   // fragment lane-chunks, 32B each
  v8f acc[8];
  #pragma unroll
  for (int c = 0; c < 8; ++c) {
    #pragma unroll
    for (int g = 0; g < 8; ++g) acc[c][g] = 0.f;
  }

  #pragma unroll
  for (int kb = 0; kb < 4; ++kb) {
    const bf16_t* ar = &sA[(wave * 16 + m) * ASTRIDE + kb * 32];
    const v8bf alo = *(const v8bf*)(ar + 8 * half);
    const v8bf ahi = *(const v8bf*)(ar + 16 + 8 * half);
    const v16bf af = cat8(alo, ahi);
    #pragma unroll
    for (int c = 0; c < 8; ++c) {
      const v16bf bfr = wfrag[(kb * 8 + c) * 32 + lane];
      acc[c] = __builtin_amdgcn_wmma_f32_16x16x32_bf16(
          false, af, false, bfr, (short)0, acc[c], false, false);
    }
  }

  #pragma unroll
  for (int c = 0; c < 8; ++c) {
    const int col = c * 16 + m;
    const float bv = bias ? bias[col] : 0.f;
    #pragma unroll
    for (int g = 0; g < 8; ++g) {
      const int row = half ? (g + 8) : g;
      float v = acc[c][g] + bv;
      if (act == 1) v = lrelu_f(v); else if (act == 2) v = fmaxf(v, 0.f);
      C[(rowbase + wave * 16 + row) * DD + col] = v;
    }
  }
}

// ---------------- flash cross-attention ----------------
// O[i] = sum_j softmax_j(q_i . k_j / sqrt(D)) * kv_j   per graph of length L.
// grid = (L/64, B); block = 128 (4 waves x 16 query rows).
__global__ __launch_bounds__(128) void k_cross_attn(
    const float* __restrict__ Q, const float* __restrict__ KV,
    float* __restrict__ O, int L)
{
  __shared__ __align__(16) bf16_t sKVb[32 * ASTRIDE];   // row-major bf16 KV tile (S frags)
  __shared__ __align__(16) bf16_t sKVt[DD * TSTRIDE];   // transposed bf16 KV tile (V frags)
  __shared__ float  sS[4][16 * 32];
  __shared__ __align__(16) bf16_t sP[4][16 * TSTRIDE];
  __shared__ float  sAl[4][16];
  __shared__ float  sL[4][16];

  const int tid = threadIdx.x;
  const int wave = tid >> 5, lane = tid & 31, half = lane >> 4, m = lane & 15;
  const int b = blockIdx.y;
  const long gq0 = (long)b * L + (long)blockIdx.x * 64 + wave * 16;
  const float scale = 0.08838834764831845f;   // 1/sqrt(128)

  // load 16x128 Q tile as 4 A-fragments (vector f32 loads -> packed bf16 cvt)
  const float* qr = Q + (gq0 + m) * DD;
  v16bf aq[4];
  #pragma unroll
  for (int kb = 0; kb < 4; ++kb) {
    const float4 a0 = *(const float4*)(qr + kb * 32 + 8 * half);
    const float4 a1 = *(const float4*)(qr + kb * 32 + 8 * half + 4);
    const float4 a2 = *(const float4*)(qr + kb * 32 + 16 + 8 * half);
    const float4 a3 = *(const float4*)(qr + kb * 32 + 16 + 8 * half + 4);
    aq[kb][0]  = (bf16_t)a0.x; aq[kb][1]  = (bf16_t)a0.y;
    aq[kb][2]  = (bf16_t)a0.z; aq[kb][3]  = (bf16_t)a0.w;
    aq[kb][4]  = (bf16_t)a1.x; aq[kb][5]  = (bf16_t)a1.y;
    aq[kb][6]  = (bf16_t)a1.z; aq[kb][7]  = (bf16_t)a1.w;
    aq[kb][8]  = (bf16_t)a2.x; aq[kb][9]  = (bf16_t)a2.y;
    aq[kb][10] = (bf16_t)a2.z; aq[kb][11] = (bf16_t)a2.w;
    aq[kb][12] = (bf16_t)a3.x; aq[kb][13] = (bf16_t)a3.y;
    aq[kb][14] = (bf16_t)a3.z; aq[kb][15] = (bf16_t)a3.w;
  }
  v8f acc[8];
  #pragma unroll
  for (int c = 0; c < 8; ++c) {
    #pragma unroll
    for (int g = 0; g < 8; ++g) acc[c][g] = 0.f;
  }
  float mrow = -1e30f, lrow = 0.f;  // valid in lanes 0..15 (one row per lane)

  for (int jb = 0; jb < L; jb += 32) {
    __syncthreads();
    const float* kvb = KV + ((long)b * L + jb) * DD;
    for (int i = tid; i < 32 * DD / 4; i += 128) {
      const float4 v = ((const float4*)kvb)[i];
      const int r = (i * 4) >> 7, col = (i * 4) & 127;
      const bf16_t b0 = (bf16_t)v.x, b1 = (bf16_t)v.y, b2 = (bf16_t)v.z, b3 = (bf16_t)v.w;
      v4bf pk; pk[0] = b0; pk[1] = b1; pk[2] = b2; pk[3] = b3;
      *(v4bf*)&sKVb[r * ASTRIDE + col] = pk;
      sKVt[(col + 0) * TSTRIDE + r] = b0;
      sKVt[(col + 1) * TSTRIDE + r] = b1;
      sKVt[(col + 2) * TSTRIDE + r] = b2;
      sKVt[(col + 3) * TSTRIDE + r] = b3;
    }
    __syncthreads();

    // S block: 16 x 32 = two 16x16 WMMA tiles; B[k][n] = KV[jt*16+n][kb*32+k]
    #pragma unroll
    for (int jt = 0; jt < 2; ++jt) {
      v8f s;
      #pragma unroll
      for (int g = 0; g < 8; ++g) s[g] = 0.f;
      #pragma unroll
      for (int kb = 0; kb < 4; ++kb) {
        const bf16_t* kr = &sKVb[(jt * 16 + m) * ASTRIDE + kb * 32];
        const v8bf blo = *(const v8bf*)(kr + 8 * half);
        const v8bf bhi = *(const v8bf*)(kr + 16 + 8 * half);
        s = __builtin_amdgcn_wmma_f32_16x16x32_bf16(
            false, aq[kb], false, cat8(blo, bhi), (short)0, s, false, false);
      }
      #pragma unroll
      for (int g = 0; g < 8; ++g) {
        const int row = half ? (g + 8) : g;
        sS[wave][row * 32 + jt * 16 + m] = s[g] * scale;
      }
    }
    __syncthreads();

    // online softmax, one row per lane (lanes 0..15)
    if (lane < 16) {
      const int r = lane;
      float mt = -1e30f;
      for (int c = 0; c < 32; ++c) mt = fmaxf(mt, sS[wave][r * 32 + c]);
      const float mnew = fmaxf(mrow, mt);
      const float al = __expf(mrow - mnew);
      float ss = 0.f;
      for (int c = 0; c < 32; ++c) {
        const float ev = __expf(sS[wave][r * 32 + c] - mnew);
        sP[wave][r * TSTRIDE + c] = (bf16_t)ev;
        ss += ev;
      }
      lrow = lrow * al + ss;
      mrow = mnew;
      sAl[wave][r] = al;
    }
    __syncthreads();

    // rescale accumulators
    #pragma unroll
    for (int g = 0; g < 8; ++g) {
      const float al = sAl[wave][half ? (g + 8) : g];
      #pragma unroll
      for (int c = 0; c < 8; ++c) acc[c][g] *= al;
    }
    // O += P(16x32) @ V(32x128); V frag from transposed tile (contiguous in j)
    const bf16_t* pr = &sP[wave][m * TSTRIDE];
    const v8bf plo = *(const v8bf*)(pr + 8 * half);
    const v8bf phi = *(const v8bf*)(pr + 16 + 8 * half);
    const v16bf ap = cat8(plo, phi);
    #pragma unroll
    for (int c = 0; c < 8; ++c) {
      const bf16_t* vr = &sKVt[(c * 16 + m) * TSTRIDE];
      const v8bf vlo = *(const v8bf*)(vr + 8 * half);
      const v8bf vhi = *(const v8bf*)(vr + 16 + 8 * half);
      acc[c] = __builtin_amdgcn_wmma_f32_16x16x32_bf16(
          false, ap, false, cat8(vlo, vhi), (short)0, acc[c], false, false);
    }
  }

  __syncthreads();
  if (lane < 16) sL[wave][lane] = lrow;
  __syncthreads();
  #pragma unroll
  for (int c = 0; c < 8; ++c) {
    #pragma unroll
    for (int g = 0; g < 8; ++g) {
      const int row = half ? (g + 8) : g;
      O[(gq0 + row) * DD + c * 16 + m] = acc[c][g] / sL[wave][row];
    }
  }
}

// ---------------- segment / scatter kernels ----------------
__global__ void k_seg_scalar(const float* __restrict__ w, const int* __restrict__ idx,
                             float* __restrict__ out, int n) {
  int p = blockIdx.x * 256 + threadIdx.x;
  if (p < n) atomicAdd(&out[idx[p]], w[p]);
}

// out[sidx[p]] += pw[p]*cA[gidx[p]]*cB[sidx[p]] * feat[gidx[p]]  (128 dims, float4)
__global__ void k_pair_scatter(const float* __restrict__ feat,
                               const int* __restrict__ gidx, const int* __restrict__ sidx,
                               const float* __restrict__ pw,
                               const float* __restrict__ cA, const float* __restrict__ cB,
                               float* __restrict__ out, int n) {
  long t = (long)blockIdx.x * 256 + threadIdx.x;
  int p = (int)(t >> 5), lane = (int)(t & 31);
  if (p >= n) return;
  float w = pw[p];
  const int g = gidx[p], s = sidx[p];
  if (cA) w *= cA[g];
  if (cB) w *= cB[s];
  if (w == 0.f) return;
  const float4 v = *(const float4*)(feat + (long)g * DD + lane * 4);
  float* o = out + (long)s * DD + lane * 4;
  atomicAdd(o + 0, v.x * w);
  atomicAdd(o + 1, v.y * w);
  atomicAdd(o + 2, v.z * w);
  atomicAdd(o + 3, v.w * w);
}

// ---------------- elementwise transforms ----------------
__global__ void k_rsqrt_arr(float* __restrict__ d, const float* __restrict__ s, int n) {
  int i = blockIdx.x * 256 + threadIdx.x;
  if (i < n) d[i] = rsqrtf(s[i]);
}
__global__ void k_rsqrt_max(float* a, int n) {
  int i = blockIdx.x * 256 + threadIdx.x;
  if (i < n) a[i] = rsqrtf(fmaxf(a[i], 1e-6f));
}
__global__ void k_recip_max(float* a, int n) {
  int i = blockIdx.x * 256 + threadIdx.x;
  if (i < n) a[i] = 1.f / fmaxf(a[i], 1e-6f);
}
__global__ void k_selfadd_lrelu(float* __restrict__ f, const float* __restrict__ h,
                                const float* __restrict__ deg, int n) {
  int i = blockIdx.x * 256 + threadIdx.x;
  if (i < n) f[i] = lrelu_f(f[i] + h[i] / deg[i >> 7]);
}
__global__ void k_scale_lrelu(float* __restrict__ f, const float* __restrict__ dv, int n) {
  int i = blockIdx.x * 256 + threadIdx.x;
  if (i < n) f[i] = lrelu_f(f[i] * dv[i >> 7]);
}
__global__ void k_div_rows(float* __restrict__ m, const float* __restrict__ cnt, int n) {
  int i = blockIdx.x * 256 + threadIdx.x;
  if (i < n) m[i] /= fmaxf(cnt[i >> 7], 1e-6f);
}
__global__ void k_xnew(float* __restrict__ e, const float* __restrict__ hef,
                       const float* __restrict__ score, const int* __restrict__ kept) {
  const int j = blockIdx.x, d = threadIdx.x;
  const int s = kept[j];
  e[(long)j * DD + d] = hef[(long)s * DD + d] * sigm_f(score[s]);
}

// ---------------- incidence structure ----------------
__global__ void k_build_pairs(const int* __restrict__ src, const int* __restrict__ dst,
                              const float* __restrict__ ea,
                              int* __restrict__ ni, int* __restrict__ hi,
                              float* __restrict__ pw) {
  int i = blockIdx.x * 256 + threadIdx.x;
  if (i < NNODES)      { ni[i] = i; hi[i] = i; pw[i] = 1.f; }
  else if (i < NPAIRS) { int e = i - NNODES; ni[i] = src[e]; hi[i] = dst[e]; pw[i] = ea[e]; }
}
__global__ void k_remap_pairs(int* __restrict__ ni, int* __restrict__ hi,
                              float* __restrict__ pw, const int* __restrict__ newid, int n) {
  int i = blockIdx.x * 256 + threadIdx.x;
  if (i >= n) return;
  const int on = ni[i], oh = hi[i];
  const int t = newid[oh];
  ni[i] = t < 0 ? 0 : t;
  hi[i] = on;
  if (t < 0) pw[i] = 0.f;
}

// ---------------- per-graph top-k via bitonic sort in LDS ----------------
__global__ __launch_bounds__(1024) void k_topk(const float* __restrict__ score,
                                               int* __restrict__ kept, int* __restrict__ newid,
                                               int npg, int k) {
  __shared__ float sv[4096];
  __shared__ int   si[4096];
  const int b = blockIdx.x;
  for (int i = threadIdx.x; i < npg; i += 1024) { sv[i] = score[b * npg + i]; si[i] = i; }
  __syncthreads();
  for (int ks = 2; ks <= npg; ks <<= 1) {
    for (int js = ks >> 1; js > 0; js >>= 1) {
      for (int i = threadIdx.x; i < npg; i += 1024) {
        const int j = i ^ js;
        if (j > i) {
          const bool up = ((i & ks) == 0);   // descending segment
          const float vi = sv[i], vj = sv[j];
          const int ii = si[i], ij = si[j];
          const bool iFirst = (vi > vj) || (vi == vj && ii < ij);
          if (up ? !iFirst : iFirst) { sv[i] = vj; sv[j] = vi; si[i] = ij; si[j] = ii; }
        }
      }
      __syncthreads();
    }
  }
  for (int j = threadIdx.x; j < k; j += 1024) {
    const int g = b * npg + si[j];
    kept[b * k + j] = g;
    newid[g] = b * k + j;
  }
}

// ---------------- score = hef . p / ||p|| ----------------
__global__ void k_pnorm(const float* __restrict__ p, float* __restrict__ out) {
  __shared__ float red[128];
  const float v = p[threadIdx.x];
  red[threadIdx.x] = v * v;
  __syncthreads();
  for (int s = 64; s > 0; s >>= 1) {
    if (threadIdx.x < s) red[threadIdx.x] += red[threadIdx.x + s];
    __syncthreads();
  }
  if (threadIdx.x == 0) out[0] = sqrtf(red[0]);
}
__global__ __launch_bounds__(256) void k_score(const float* __restrict__ hef,
                                               const float* __restrict__ p,
                                               const float* __restrict__ pn,
                                               float* __restrict__ score, int nhe) {
  const int w = blockIdx.x * 8 + (threadIdx.x >> 5);
  const int lane = threadIdx.x & 31;
  if (w >= nhe) return;
  const float4 hv = *(const float4*)(hef + (long)w * DD + lane * 4);
  const float4 pv = *(const float4*)(p + lane * 4);
  float s = hv.x*pv.x + hv.y*pv.y + hv.z*pv.z + hv.w*pv.w;
  for (int off = 16; off > 0; off >>= 1) s += __shfl_xor(s, off, 32);
  if (lane == 0) score[w] = s / pn[0];
}

// ---------------- attention readout ----------------
__global__ __launch_bounds__(128) void k_readout(const float* __restrict__ x,
                                                 const float* __restrict__ Wr,
                                                 float* __restrict__ scores, int n, int off) {
  __shared__ float mean[128], ctx[128];
  __shared__ float coef[4096];
  const int b = blockIdx.x, d = threadIdx.x;
  const float* xb = x + (long)b * n * DD;
  float s = 0.f;
  for (int r = 0; r < n; ++r) s += xb[(long)r * DD + d];
  mean[d] = s / (float)n;
  __syncthreads();
  float c = 0.f;
  for (int e = 0; e < DD; ++e) c += mean[e] * Wr[e * DD + d];
  ctx[d] = tanhf(c);
  __syncthreads();
  for (int r = d; r < n; r += 128) {
    float t = 0.f;
    const float* row = xb + (long)r * DD;
    for (int e = 0; e < DD; ++e) t += row[e] * ctx[e];
    coef[r] = sigm_f(t);
  }
  __syncthreads();
  float acc = 0.f;
  for (int r = 0; r < n; ++r) acc += coef[r] * xb[(long)r * DD + d];
  scores[b * 1024 + off + d] = acc;
}

// ---------------- final MLP head ----------------
__global__ __launch_bounds__(512) void k_mlp(const float* __restrict__ scores,
                                             const float* __restrict__ Wm1, const float* __restrict__ bm1,
                                             const float* __restrict__ Wm2, const float* __restrict__ bm2,
                                             const float* __restrict__ Wm3, const float* __restrict__ bm3,
                                             float* __restrict__ out) {
  __shared__ float sc[8 * 1024];
  __shared__ float h1[8 * 512];
  __shared__ float h2[8 * 128];
  const int t = threadIdx.x;
  for (int i = t; i < 8192; i += 512) sc[i] = scores[i];
  __syncthreads();
  for (int o = t; o < 8 * 512; o += 512) {
    const int b = o >> 9, j = o & 511;
    float a = bm1[j];
    for (int e = 0; e < 1024; ++e) a += sc[b * 1024 + e] * Wm1[e * 512 + j];
    h1[o] = fmaxf(a, 0.f);
  }
  __syncthreads();
  for (int o = t; o < 8 * 128; o += 512) {
    const int b = o >> 7, j = o & 127;
    float a = bm2[j];
    for (int e = 0; e < 512; ++e) a += h1[b * 512 + e] * Wm2[e * 128 + j];
    h2[o] = fmaxf(a, 0.f);
  }
  __syncthreads();
  if (t < 8) {
    float a = bm3[0];
    for (int e = 0; e < 128; ++e) a += h2[t * 128 + e] * Wm3[e];
    out[t] = sigm_f(a);
  }
}

// =====================================================================
static inline int cdiv_i(int a, int b) { return (a + b - 1) / b; }

extern "C" void kernel_launch(void* const* d_in, const int* in_sizes, int n_in,
                              void* d_out, int out_size, void* d_ws, size_t ws_size,
                              hipStream_t stream) {
  (void)in_sizes; (void)n_in; (void)out_size; (void)ws_size;
  const float* x_[2]  = {(const float*)d_in[0], (const float*)d_in[1]};
  const float* ea_[2] = {(const float*)d_in[2], (const float*)d_in[3]};
  const float* Wg  = (const float*)d_in[4];
  const float* bg  = (const float*)d_in[5];
  const float* Wl_[3] = {(const float*)d_in[6], (const float*)d_in[8],  (const float*)d_in[10]};
  const float* bl_[3] = {(const float*)d_in[7], (const float*)d_in[9],  (const float*)d_in[11]};
  const float* Wc_[3] = {(const float*)d_in[12], (const float*)d_in[13], (const float*)d_in[14]};
  const float* Wr_[4] = {(const float*)d_in[15], (const float*)d_in[16],
                         (const float*)d_in[17], (const float*)d_in[18]};
  const float* pv_[3] = {(const float*)d_in[19], (const float*)d_in[20], (const float*)d_in[21]};
  const float* Wm1 = (const float*)d_in[22]; const float* bm1 = (const float*)d_in[23];
  const float* Wm2 = (const float*)d_in[24]; const float* bm2 = (const float*)d_in[25];
  const float* Wm3 = (const float*)d_in[26]; const float* bm3 = (const float*)d_in[27];
  const int* src_[2] = {(const int*)d_in[28], (const int*)d_in[30]};
  const int* dst_[2] = {(const int*)d_in[29], (const int*)d_in[31]};
  float* outp = (float*)d_out;

  // ---- workspace layout ----
  char* base = (char*)d_ws;
  size_t off = 0;
  auto alloc = [&](size_t nbytes) -> char* {
    char* p = base + off;
    off += (nbytes + 255) & ~(size_t)255;
    return p;
  };
  const size_t NF = (size_t)NNODES * DD * 4;
  float* f_[2] = {(float*)alloc(NF), (float*)alloc(NF)};
  float* t_[2] = {(float*)alloc(NF), (float*)alloc(NF)};
  float* m_[2] = {(float*)alloc(NF), (float*)alloc(NF)};
  float* e_[2] = {(float*)alloc((size_t)16384 * DD * 4), (float*)alloc((size_t)16384 * DD * 4)};
  float* dv_[2] = {(float*)alloc((size_t)NNODES * 4), (float*)alloc((size_t)NNODES * 4)};
  float* bi_[2] = {(float*)alloc((size_t)NNODES * 4), (float*)alloc((size_t)NNODES * 4)};
  float* sco_[2] = {(float*)alloc((size_t)NNODES * 4), (float*)alloc((size_t)NNODES * 4)};
  int* ni_[2] = {(int*)alloc((size_t)NPAIRS * 4), (int*)alloc((size_t)NPAIRS * 4)};
  int* hi_[2] = {(int*)alloc((size_t)NPAIRS * 4), (int*)alloc((size_t)NPAIRS * 4)};
  float* pw_[2] = {(float*)alloc((size_t)NPAIRS * 4), (float*)alloc((size_t)NPAIRS * 4)};
  int* newid = (int*)alloc((size_t)NNODES * 4);
  int* kept  = (int*)alloc((size_t)16384 * 4);
  float* scoresv = (float*)alloc((size_t)8192 * 4);
  float* pn = (float*)alloc(256);
  bf16_t* wb[7];
  for (int i = 0; i < 7; ++i) wb[i] = (bf16_t*)alloc((size_t)DD * DD * 2);

  // ---- small launch helpers ----
  auto fillf = [&](float* p, float v, int n) {
    k_fill_f32<<<cdiv_i(n, 256), 256, 0, stream>>>(p, v, n);
  };
  auto filli = [&](int* p, int v, int n) {
    k_fill_i32<<<cdiv_i(n, 256), 256, 0, stream>>>(p, v, n);
  };
  auto gemm = [&](const float* A, const bf16_t* W, const float* bias, float* C, int M, int act) {
    k_gemm128<<<M / 64, 128, 0, stream>>>(A, W, bias, C, M, act);
  };
  auto pscat = [&](const float* feat, const int* gi, const int* si, const float* pw,
                   const float* cA, const float* cB, float* out, int n) {
    long thr = (long)n * 32;
    k_pair_scatter<<<(int)((thr + 255) / 256), 256, 0, stream>>>(feat, gi, si, pw, cA, cB, out, n);
  };
  auto segs = [&](const float* w, const int* idx, float* out, int n) {
    k_seg_scalar<<<cdiv_i(n, 256), 256, 0, stream>>>(w, idx, out, n);
  };

  // ---- pack GEMM weights to bf16 fragment order (Wg, W1..3, Wc1..3) ----
  const float* wsrc[7] = {Wg, Wl_[0], Wl_[1], Wl_[2], Wc_[0], Wc_[1], Wc_[2]};
  for (int i = 0; i < 7; ++i)
    k_packW<<<cdiv_i(DD * DD, 256), 256, 0, stream>>>(wsrc[i], wb[i]);

  // ================= GCN layer + readout s0 + incidence build =================
  for (int s = 0; s < 2; ++s) {
    gemm(x_[s], wb[0], bg, t_[s], NNODES, 0);                 // h = x@Wg + bg
    fillf(dv_[s], 1.f, NNODES);                               // deg = 1 (self loop)
    segs(ea_[s], dst_[s], dv_[s], NEDGES);                    // deg += segsum(w, dst)
    k_rsqrt_arr<<<cdiv_i(NNODES, 256), 256, 0, stream>>>(bi_[s], dv_[s], NNODES);  // inv
    fillf(f_[s], 0.f, NNODES * DD);
    pscat(t_[s], src_[s], dst_[s], ea_[s], bi_[s], bi_[s], f_[s], NEDGES);
    k_selfadd_lrelu<<<cdiv_i(NNODES * DD, 256), 256, 0, stream>>>(f_[s], t_[s], dv_[s], NNODES * DD);
    k_readout<<<BGR, 128, 0, stream>>>(f_[s], Wr_[0], scoresv, NNODES / BGR, 0 * 256 + s * 128);
    k_build_pairs<<<cdiv_i(NPAIRS, 256), 256, 0, stream>>>(src_[s], dst_[s], ea_[s],
                                                           ni_[s], hi_[s], pw_[s]);
  }

  // ================= 3 stages of hconv -> hpool -> cross -> readout ==========
  int nv = NNODES, nhe = NNODES;
  const int kk[3] = {2048, 2048, 1024};
  for (int st = 0; st < 3; ++st) {
    const int k = kk[st];
    const int npg = nhe / BGR;
    // ---- hconv per side ----
    for (int s = 0; s < 2; ++s) {
      gemm(f_[s], wb[1 + st], bl_[st], t_[s], nv, 0);         // h = x@W + b
      fillf(dv_[s], 0.f, nv);
      segs(pw_[s], ni_[s], dv_[s], NPAIRS);
      k_rsqrt_max<<<cdiv_i(nv, 256), 256, 0, stream>>>(dv_[s], nv);
      fillf(bi_[s], 0.f, nhe);
      segs(pw_[s], hi_[s], bi_[s], NPAIRS);
      k_recip_max<<<cdiv_i(nhe, 256), 256, 0, stream>>>(bi_[s], nhe);
      fillf(m_[s], 0.f, nhe * DD);
      pscat(t_[s], ni_[s], hi_[s], pw_[s], dv_[s], nullptr, m_[s], NPAIRS);
      fillf(f_[s], 0.f, nv * DD);
      pscat(m_[s], hi_[s], ni_[s], pw_[s], bi_[s], nullptr, f_[s], NPAIRS);
      k_scale_lrelu<<<cdiv_i(nv * DD, 256), 256, 0, stream>>>(f_[s], dv_[s], nv * DD);
    }
    // ---- hpool per side ----
    k_pnorm<<<1, 128, 0, stream>>>(pv_[st], pn);
    for (int s = 0; s < 2; ++s) {
      fillf(bi_[s], 0.f, nhe);                                // cnt (raw)
      segs(pw_[s], hi_[s], bi_[s], NPAIRS);
      fillf(m_[s], 0.f, nhe * DD);
      pscat(f_[s], ni_[s], hi_[s], pw_[s], nullptr, nullptr, m_[s], NPAIRS);
      k_div_rows<<<cdiv_i(nhe * DD, 256), 256, 0, stream>>>(m_[s], bi_[s], nhe * DD); // hef
      k_score<<<cdiv_i(nhe, 8), 256, 0, stream>>>(m_[s], pv_[st], pn, sco_[s], nhe);
      filli(newid, -1, nhe);
      k_topk<<<BGR, 1024, 0, stream>>>(sco_[s], kept, newid, npg, k);
      k_xnew<<<BGR * k, 128, 0, stream>>>(e_[s], m_[s], sco_[s], kept);
      k_remap_pairs<<<cdiv_i(NPAIRS, 256), 256, 0, stream>>>(ni_[s], hi_[s], pw_[s], newid, NPAIRS);
    }
    const int nv_new = BGR * k;
    const int nhe_new = nv;
    nv = nv_new; nhe = nhe_new;
    // ---- cross attention (flash) + Wc projection + readout ----
    const int L = k;
    dim3 agrid(L / 64, BGR);
    k_cross_attn<<<agrid, 128, 0, stream>>>(e_[0], e_[1], t_[0], L);
    k_cross_attn<<<agrid, 128, 0, stream>>>(e_[1], e_[0], t_[1], L);
    gemm(t_[0], wb[4 + st], nullptr, f_[0], nv, 2);           // relu(attn @ Wc)
    gemm(t_[1], wb[4 + st], nullptr, f_[1], nv, 2);
    k_readout<<<BGR, 128, 0, stream>>>(f_[0], Wr_[1 + st], scoresv, k, (st + 1) * 256 + 0);
    k_readout<<<BGR, 128, 0, stream>>>(f_[1], Wr_[1 + st], scoresv, k, (st + 1) * 256 + 128);
  }

  // ================= MLP head =================
  k_mlp<<<1, 512, 0, stream>>>(scoresv, Wm1, bm1, Wm2, bm2, Wm3, bm3, outp);
}
